// BrainNetGIN_64811056497272
// MI455X (gfx1250) — matrix operand order repaired and checked by reading, not compiled, not run on hardware
//
#include <hip/hip_runtime.h>

// ---------------------------------------------------------------------------
// BrainNet GIN for MI455X (gfx1250, wave32, WMMA).
//
//  - Edge aggregation: one workgroup per (graph, 128-col chunk); 400x132 f32
//    tile in LDS (stride 132 breaks 64-bank conflicts), ds_add_f32 atomics.
//  - GEMMs: v_wmma_f32_16x16x32_f16, f16 operands / f32 accumulate.
//    K is a template parameter (416 / 128) so the k-loop fully unrolls:
//    all fragment loads use immediate offsets (A: global_load_b128 clauses) and
//    the scheduler pipelines loads across WMMAs instead of wait-0 per tile.
//    Fragment layouts per CDNA5 ISA 7.12.2:
//      A 16x32 f16 : lane&15 = M row; halves 0..7 -> k = 8*(lane>>4)+0..7,
//                    halves 8..15 -> k = 16+8*(lane>>4)+0..7  (2x16B loads)
//      B 32x16 f16 : lane&15 = N col; halves -> same k as A  (16 strided 2B loads)
//      C 16x16 f32 : element (m = vgpr + 8*(lane>>4), n = lane&15)
//  - BatchNorm (training-mode stats over all 25600 nodes) as
//    stats -> finalize -> apply(+ReLU, f16 convert) around GEMM-a.
// ---------------------------------------------------------------------------

typedef _Float16 half_t;
typedef __attribute__((ext_vector_type(16))) _Float16 v16h;
typedef __attribute__((ext_vector_type(8)))  float    v8f;

#define NNODES 25600
#define FIN    400
#define HDIM   128
#define NEDGE  512000
#define NGRAPH 64
#define NPG    400
#define EPG    8000
#define K0PAD  416          // 404 features padded to multiple of 32 (zeros)
#define LDS_STRIDE 132      // 400 x 132 f32 tile, conflict-free banks
#define BN_EPS 1e-5f

// ---------------- feature build: h0 = [x | ge[group] | he[n%2] | 0-pad] -----
__global__ __launch_bounds__(256) void build_h0(
    const float* __restrict__ x, const int* __restrict__ group_ids,
    const float* __restrict__ ge, const float* __restrict__ he,
    half_t* __restrict__ h0)
{
    int idx = blockIdx.x * 256 + threadIdx.x;
    if (idx >= NNODES * K0PAD) return;
    int n = idx / K0PAD;
    int k = idx - n * K0PAD;
    float v;
    if (k < FIN)            v = x[(size_t)n * FIN + k];
    else if (k < FIN + 2)   v = ge[group_ids[n] * 2 + (k - FIN)];
    else if (k < FIN + 4)   v = he[(n & 1) * 2 + (k - FIN - 2)];
    else                    v = 0.0f;
    h0[idx] = (half_t)v;
}

// ---------------- weight f32 -> f16 with zero row padding -------------------
__global__ __launch_bounds__(256) void cvt_weight(
    const float* __restrict__ w, half_t* __restrict__ w16, int Kin, int Kpad)
{
    int idx = blockIdx.x * 256 + threadIdx.x;
    if (idx >= Kpad * HDIM) return;
    int k = idx / HDIM;
    w16[idx] = (half_t)(k < Kin ? w[idx] : 0.0f);
}

// ---------------- GIN aggregate: z = h + sum_{(s,d)} h[s]  (per graph) ------
__global__ __launch_bounds__(256) void gin_aggregate(
    const half_t* __restrict__ h, int ldh,
    const int* __restrict__ srcIdx, const int* __restrict__ dstIdx,
    half_t* __restrict__ z)
{
    extern __shared__ float smem[];               // 400 * LDS_STRIDE floats
    const int g    = blockIdx.x;
    const int c0   = blockIdx.y * 128;
    const int CH   = (ldh - c0 < 128) ? (ldh - c0) : 128;   // 128 or 32
    const int base = g * NPG;

    // init LDS tile with h (f16 -> f32)
    for (int i = threadIdx.x; i < NPG * CH; i += 256) {
        int n = i / CH, c = i - n * CH;
        smem[n * LDS_STRIDE + c] = (float)h[(size_t)(base + n) * ldh + c0 + c];
    }
    __syncthreads();

    // scatter-add h[src] into dst rows (LDS float atomics -> ds_add_f32)
    const int eEnd = (g + 1) * EPG;
    for (int e = g * EPG + threadIdx.x; e < eEnd; e += 256) {
        const int s  = srcIdx[e];
        const int dl = dstIdx[e] - base;
        const half_t* hs   = h + (size_t)s * ldh + c0;
        float*        zrow = smem + dl * LDS_STRIDE;
        for (int c = 0; c < CH; c += 8) {
            uint4 u = *(const uint4*)(hs + c);            // 8 halves, 16B aligned
            const half_t* hv = (const half_t*)&u;
#pragma unroll
            for (int j = 0; j < 8; ++j)
                __hip_atomic_fetch_add(&zrow[c + j], (float)hv[j],
                                       __ATOMIC_RELAXED, __HIP_MEMORY_SCOPE_WORKGROUP);
        }
    }
    __syncthreads();

    // write z back as f16
    for (int i = threadIdx.x; i < NPG * CH; i += 256) {
        int n = i / CH, c = i - n * CH;
        z[(size_t)(base + n) * ldh + c0 + c] = (half_t)smem[n * LDS_STRIDE + c];
    }
}

// ---------------- WMMA GEMM: out = A[M,K(f16)] @ B[K,128(f16)] + bias -------
// block = 256 threads = 8 waves; block tile = 128 rows x 128 cols;
// wave w owns 16-col strip, loops 8 row tiles; K stepped by 32, full unroll.
template <int K>
__global__ __launch_bounds__(256) void gemm_wmma(
    const half_t* __restrict__ A,
    const half_t* __restrict__ Bw,
    const float* __restrict__ bias,
    float* __restrict__ outF, half_t* __restrict__ outH, int relu)
{
    const int wave = threadIdx.x >> 5;
    const int lane = threadIdx.x & 31;
    const int m0   = blockIdx.x * 128;
    const int n0   = wave * 16;
    const int grp  = lane >> 4;
    const int l16  = lane & 15;

    // base pointers; all k-dependent addressing folds into immediate offsets
    const half_t* arow[8];
#pragma unroll
    for (int t = 0; t < 8; ++t)
        arow[t] = A + (size_t)(m0 + t * 16 + l16) * K + grp * 8;
    const half_t* brow = Bw + (size_t)(grp * 8) * HDIM + n0 + l16;

    v8f acc[8] = {};

#pragma unroll
    for (int k0 = 0; k0 < K; k0 += 32) {
        // B fragment: lane&15 = n col; halves s -> k = 8*(lane>>4)+s (s<8), 16+8*(lane>>4)+s-8 (s>=8)
        v16h bf;
#pragma unroll
        for (int s = 0; s < 16; ++s)
            bf[s] = brow[(size_t)(k0 + (s < 8 ? s : s + 8)) * HDIM];

        // batch all 8 A fragments (one load clause), then 8 back-to-back WMMAs
        v16h af[8];
#pragma unroll
        for (int t = 0; t < 8; ++t) {
            ((uint4*)&af[t])[0] = *(const uint4*)(arow[t] + k0);
            ((uint4*)&af[t])[1] = *(const uint4*)(arow[t] + k0 + 16);
        }
#pragma unroll
        for (int t = 0; t < 8; ++t)
            acc[t] = __builtin_amdgcn_wmma_f32_16x16x32_f16(
                false, af[t], false, bf, (short)0, acc[t], false, false);
    }

    const int   nc   = n0 + l16;
    const float bval = bias[nc];
#pragma unroll
    for (int t = 0; t < 8; ++t) {
#pragma unroll
        for (int r = 0; r < 8; ++r) {
            int   m = m0 + t * 16 + r + grp * 8;
            float v = acc[t][r] + bval;
            if (relu) v = v > 0.0f ? v : 0.0f;
            if (outF) outF[(size_t)m * HDIM + nc] = v;
            if (outH) outH[(size_t)m * HDIM + nc] = (half_t)v;
        }
    }
}

// ---------------- BatchNorm pieces ------------------------------------------
__global__ __launch_bounds__(256) void zero_f32(float* p, int n)
{
    int i = blockIdx.x * 256 + threadIdx.x;
    if (i < n) p[i] = 0.0f;
}

// 128 threads = 128 columns; each block reduces 256 rows; coalesced rows.
__global__ __launch_bounds__(128) void bn_stats(
    const float* __restrict__ y, float* __restrict__ stats)
{
    const int c  = threadIdx.x;
    const int r0 = blockIdx.x * 256;
    float s = 0.0f, s2 = 0.0f;
    for (int r = r0; r < r0 + 256; ++r) {
        float v = y[(size_t)r * HDIM + c];
        s += v; s2 += v * v;
    }
    __hip_atomic_fetch_add(&stats[c],        s,  __ATOMIC_RELAXED, __HIP_MEMORY_SCOPE_AGENT);
    __hip_atomic_fetch_add(&stats[HDIM + c], s2, __ATOMIC_RELAXED, __HIP_MEMORY_SCOPE_AGENT);
}

__global__ __launch_bounds__(128) void bn_finalize(
    float* __restrict__ stats, const float* __restrict__ gamma,
    const float* __restrict__ beta, float inv_n)
{
    const int c    = threadIdx.x;
    float mean = stats[c] * inv_n;
    float var  = stats[HDIM + c] * inv_n - mean * mean;
    float sc   = gamma[c] * rsqrtf(var + BN_EPS);
    stats[256 + c] = sc;
    stats[384 + c] = beta[c] - mean * sc;
}

// z2 = f16(relu(y*scale + shift))
__global__ __launch_bounds__(256) void bn_apply(
    const float* __restrict__ y, const float* __restrict__ stats,
    half_t* __restrict__ z2)
{
    int i = blockIdx.x * 256 + threadIdx.x;
    if (i >= NNODES * HDIM) return;
    int   c = i & (HDIM - 1);
    float v = y[i] * stats[256 + c] + stats[384 + c];
    z2[i] = (half_t)(v > 0.0f ? v : 0.0f);
}

// ---------------- pooling + head --------------------------------------------
__global__ __launch_bounds__(128) void pool_graph(
    const half_t* __restrict__ h, float* __restrict__ pooled)
{
    const int g = blockIdx.x, c = threadIdx.x;
    const half_t* p = h + (size_t)g * NPG * HDIM + c;
    float s = 0.0f;
    for (int n = 0; n < NPG; ++n) s += (float)p[(size_t)n * HDIM];
    pooled[g * HDIM + c] = s;
}

__global__ __launch_bounds__(128) void head(
    const float* __restrict__ pooled,
    const float* __restrict__ wfa, const float* __restrict__ bfa,
    const float* __restrict__ wfb, const float* __restrict__ bfb,
    float* __restrict__ out)
{
    __shared__ float prow[HDIM];
    __shared__ float t[HDIM];
    const int g = blockIdx.x, c = threadIdx.x;
    prow[c] = pooled[g * HDIM + c];
    __syncthreads();
    float s = bfa[c];
    for (int k = 0; k < HDIM; ++k) s += prow[k] * wfa[k * HDIM + c];
    t[c] = s > 0.0f ? s : 0.0f;
    __syncthreads();
    if (c < 2) {
        float o = bfb[c];
        for (int k = 0; k < HDIM; ++k) o += t[k] * wfb[k * 2 + c];
        out[g * 2 + c] = o;
    }
}

// ---------------------------------------------------------------------------
extern "C" void kernel_launch(void* const* d_in, const int* in_sizes, int n_in,
                              void* d_out, int out_size, void* d_ws, size_t ws_size,
                              hipStream_t stream)
{
    const float* x         = (const float*)d_in[0];
    const int*   ei        = (const int*)  d_in[1];
    const int*   srcIdx    = ei;
    const int*   dstIdx    = ei + NEDGE;
    const int*   group_ids = (const int*)  d_in[3];
    const float* ge        = (const float*)d_in[4];
    const float* he        = (const float*)d_in[5];

    const float* wa[3] = { (const float*)d_in[6],  (const float*)d_in[12], (const float*)d_in[18] };
    const float* ba[3] = { (const float*)d_in[7],  (const float*)d_in[13], (const float*)d_in[19] };
    const float* gm[3] = { (const float*)d_in[8],  (const float*)d_in[14], (const float*)d_in[20] };
    const float* be[3] = { (const float*)d_in[9],  (const float*)d_in[15], (const float*)d_in[21] };
    const float* wb[3] = { (const float*)d_in[10], (const float*)d_in[16], (const float*)d_in[22] };
    const float* bb[3] = { (const float*)d_in[11], (const float*)d_in[17], (const float*)d_in[23] };
    const float* wfa = (const float*)d_in[24];
    const float* bfa = (const float*)d_in[25];
    const float* wfb = (const float*)d_in[26];
    const float* bfb = (const float*)d_in[27];

    // ---- workspace carve-out (≈63 MB) ----
    char* p = (char*)d_ws;
    auto carve = [&](size_t bytes) { char* r = p; p += (bytes + 255) & ~(size_t)255; return r; };
    half_t* hbuf  = (half_t*)carve((size_t)NNODES * K0PAD * 2);   // current h (f16)
    half_t* zbuf  = (half_t*)carve((size_t)NNODES * K0PAD * 2);   // z = h + agg (f16)
    float*  ybuf  = (float*) carve((size_t)NNODES * HDIM * 4);    // pre-BN GEMM-a out
    half_t* z2buf = (half_t*)carve((size_t)NNODES * HDIM * 2);    // post-BN ReLU (f16)
    half_t* w16   = (half_t*)carve((size_t)(K0PAD * HDIM + 5 * HDIM * HDIM) * 2);
    float*  stats = (float*) carve(512 * 4);
    float*  pooled= (float*) carve((size_t)NGRAPH * HDIM * 4);

    half_t* wa16[3], *wb16[3];
    wa16[0] = w16;
    wa16[1] = wa16[0] + K0PAD * HDIM;
    wa16[2] = wa16[1] + HDIM * HDIM;
    wb16[0] = wa16[2] + HDIM * HDIM;
    wb16[1] = wb16[0] + HDIM * HDIM;
    wb16[2] = wb16[1] + HDIM * HDIM;

    // ---- weight conversion (f32 -> f16, zero-pad K for layer 0) ----
    cvt_weight<<<(K0PAD * HDIM + 255) / 256, 256, 0, stream>>>(wa[0], wa16[0], FIN + 4, K0PAD);
    for (int L = 1; L < 3; ++L)
        cvt_weight<<<(HDIM * HDIM + 255) / 256, 256, 0, stream>>>(wa[L], wa16[L], HDIM, HDIM);
    for (int L = 0; L < 3; ++L)
        cvt_weight<<<(HDIM * HDIM + 255) / 256, 256, 0, stream>>>(wb[L], wb16[L], HDIM, HDIM);

    // ---- input features ----
    build_h0<<<(NNODES * K0PAD + 255) / 256, 256, 0, stream>>>(x, group_ids, ge, he, hbuf);

    const size_t aggLds = (size_t)NPG * LDS_STRIDE * 4;   // 211,200 B of WGP LDS

    for (int L = 0; L < 3; ++L) {
        const int ldh = (L == 0) ? K0PAD : HDIM;
        const int nch = (ldh + 127) / 128;

        gin_aggregate<<<dim3(NGRAPH, nch), 256, aggLds, stream>>>(hbuf, ldh, srcIdx, dstIdx, zbuf);

        if (L == 0)
            gemm_wmma<K0PAD><<<NNODES / 128, 256, 0, stream>>>(zbuf, wa16[L], ba[L],
                                                               ybuf, (half_t*)nullptr, 0);
        else
            gemm_wmma<HDIM><<<NNODES / 128, 256, 0, stream>>>(zbuf, wa16[L], ba[L],
                                                              ybuf, (half_t*)nullptr, 0);

        zero_f32<<<1, 256, 0, stream>>>(stats, 256);
        bn_stats<<<NNODES / 256, 128, 0, stream>>>(ybuf, stats);
        bn_finalize<<<1, 128, 0, stream>>>(stats, gm[L], be[L], 1.0f / (float)NNODES);
        bn_apply<<<(NNODES * HDIM) / 256, 256, 0, stream>>>(ybuf, stats, z2buf);

        gemm_wmma<HDIM><<<NNODES / 128, 256, 0, stream>>>(z2buf, wb16[L], bb[L],
                                                          (float*)nullptr, hbuf, 1);
    }

    pool_graph<<<NGRAPH, 128, 0, stream>>>(hbuf, pooled);
    head<<<NGRAPH, 128, 0, stream>>>(pooled, wfa, bfa, wfb, bfb, (float*)d_out);
}
